// PaiNN_44513041056429
// MI455X (gfx1250) — matrix operand order repaired and checked by reading, not compiled, not run on hardware
//
#include <hip/hip_runtime.h>
#include <hip/hip_bf16.h>
#include <math.h>

typedef _Float16 half_t;
typedef __attribute__((ext_vector_type(16))) _Float16 v16h;
typedef __attribute__((ext_vector_type(8)))  _Float16 v8h;
typedef __attribute__((ext_vector_type(8)))  float    v8f;

#define PI_F     3.14159265358979f
#define CUTOFF_F 3.0f
#define F_DIM    128

// Fast SiLU: x * sigmoid(x) with hardware v_rcp_f32 (no IEEE div chain).
__device__ __forceinline__ float silu_f(float x) {
    float e = __expf(-x);
    return x * __builtin_amdgcn_rcpf(1.0f + e);
}

__device__ __forceinline__ v16h cat8(v8h lo, v8h hi) {
    return __builtin_shufflevector(lo, hi,
        0,1,2,3,4,5,6,7,8,9,10,11,12,13,14,15);
}

// ---------------------------------------------------------------------------
// Block-cooperative WMMA GEMM:  out[M,N] = act(A[M,K] @ B[K,N] + bias) * rowscale
// A: f16 row-major [M][K].  BT: f16 = B transposed, row-major [N][K].
// Block (256 thr = 8 waves) computes a 128x64 tile: the 64xK B panel is staged
// once into LDS (shared by all 8 waves), then each wave computes one 16x64
// strip: 4 f32 accumulators, K-loop step 32, A double-buffered from global,
// B fragments from LDS (ds_load_b128 pairs), v_wmma_f32_16x16x32_f16.
// Requires: M%16==0, N%64==0 (true for all call sites here).
// ---------------------------------------------------------------------------
template<int KD>
__global__ __launch_bounds__(256)
void gemm16_wmma(const half_t* __restrict__ Ah, const half_t* __restrict__ BT,
                 const float* __restrict__ bias, const float* __restrict__ rowscale,
                 float* __restrict__ outF, half_t* __restrict__ outH,
                 int Mrows, int Ncols, int act_silu)
{
    constexpr int KP = KD + 8;                 // padded LDS row stride (halfs)
    __shared__ __align__(16) half_t ldsB[64 * KP];

    const int nBase = blockIdx.x << 6;

    // Cooperative stage of B panel BT[nBase..nBase+63][0..KD) -> LDS (v8h chunks)
    for (int idx = threadIdx.x; idx < 64 * (KD / 8); idx += 256) {
        const int col = idx / (KD / 8);
        const int kk  = (idx % (KD / 8)) * 8;
        *(v8h*)&ldsB[col * KP + kk] =
            *(const v8h*)(BT + (size_t)(nBase + col) * KD + kk);
    }
    __syncthreads();

    const int lane = threadIdx.x & 31;
    const int wave = threadIdx.x >> 5;
    const int rowTile = blockIdx.y * 8 + wave;
    if ((rowTile << 4) >= Mrows) return;        // wave-uniform: EXEC stays full
    const int rowBase = rowTile << 4;
    const int l16 = lane & 15;
    const int hi  = lane >> 4;                  // 0 or 1

    // A fragment base: lane holds row (lane&15); K-halves split by lane>=16
    const half_t* arow = Ah + (size_t)(rowBase + l16) * KD + hi * 8;

    v8f acc[4];
    #pragma unroll
    for (int t = 0; t < 4; ++t) acc[t] = (v8f){0.f,0.f,0.f,0.f,0.f,0.f,0.f,0.f};

    auto loadA = [&](int k0) -> v16h {
        v8h alo = *(const v8h*)(arow + k0);
        v8h ahi = *(const v8h*)(arow + k0 + 16);
        return cat8(alo, ahi);
    };

    v16h a_cur = loadA(0);
    #pragma unroll
    for (int k0 = 0; k0 < KD; k0 += 32) {
        v16h a_next = (k0 + 32 < KD) ? loadA(k0 + 32) : a_cur;
        if (k0 + 64 < KD)
            __builtin_prefetch(arow + k0 + 64, 0, 0);   // global_prefetch_b8
        v16h bfrag[4];
        #pragma unroll
        for (int t = 0; t < 4; ++t) {
            // lane holds col (lane&15) of subtile t, K halfs [8h,8h+8) and [16+8h,24+8h)
            const half_t* bp = &ldsB[(t * 16 + l16) * KP + k0 + hi * 8];
            v8h b0 = *(const v8h*)(bp);
            v8h b1 = *(const v8h*)(bp + 16);
            bfrag[t] = cat8(b0, b1);
        }
        #pragma unroll
        for (int t = 0; t < 4; ++t)
            acc[t] = __builtin_amdgcn_wmma_f32_16x16x32_f16(
                false, a_cur, false, bfrag[t], (short)0, acc[t], false, false);
        a_cur = a_next;
    }

    // Epilogue: C/D layout -> VGPR g, lane L: row = g + (L>=16)*8, col = L&15
    #pragma unroll
    for (int t = 0; t < 4; ++t) {
        const int col = nBase + t * 16 + l16;
        const float bc = bias ? bias[col] : 0.0f;
        #pragma unroll
        for (int g = 0; g < 8; ++g) {
            const int row = rowBase + hi * 8 + g;
            float x = acc[t][g] + bc;
            if (act_silu) x = silu_f(x);
            if (rowscale) x *= rowscale[row];
            if (outF) outF[(size_t)row * Ncols + col] = x;
            if (outH) outH[(size_t)row * Ncols + col] = (half_t)x;
        }
    }
}

// Pack W[K,N] (f32) -> WT[N][Kpad] (f16), zero-padding K up to Kpad.
__global__ void pack_bt_k(const float* __restrict__ W, half_t* __restrict__ WT,
                          int K, int N, int Kpad)
{
    int tid = blockIdx.x * blockDim.x + threadIdx.x;
    if (tid >= N * Kpad) return;
    int n = tid / Kpad, k = tid % Kpad;
    WT[(size_t)n * Kpad + k] = (k < K) ? (half_t)W[(size_t)k * N + n] : (half_t)0.0f;
}

__global__ void embed_k(const int* __restrict__ sidx, const float* __restrict__ emb,
                        float* __restrict__ s, int A)
{
    int tid = blockIdx.x * blockDim.x + threadIdx.x;
    if (tid >= A * F_DIM) return;
    int a = tid >> 7, f = tid & 127;
    s[tid] = emb[(size_t)sidx[a] * F_DIM + f];
}

__global__ void f2h_k(const float* __restrict__ x, half_t* __restrict__ y, size_t n)
{
    size_t tid = blockIdx.x * (size_t)blockDim.x + threadIdx.x;
    if (tid < n) y[tid] = (half_t)x[tid];
}

// Per-edge geometry: d, unit dir, sinc RBF basis (padded 20->32), cosine cutoff.
__global__ void edge_prep_k(const float* __restrict__ r, float* __restrict__ dirn,
                            float* __restrict__ fcut, half_t* __restrict__ rbf, int E)
{
    int e = blockIdx.x * blockDim.x + threadIdx.x;
    if (e >= E) return;
    float rx = r[3*e], ry = r[3*e+1], rz = r[3*e+2];
    float d   = sqrtf(rx*rx + ry*ry + rz*rz);
    float inv = 1.0f / d;
    dirn[3*e]   = rx * inv;
    dirn[3*e+1] = ry * inv;
    dirn[3*e+2] = rz * inv;
    float c = (PI_F / CUTOFF_F) * d;
    fcut[e] = (d < CUTOFF_F) ? 0.5f * (__cosf(c) + 1.0f) : 0.0f;
    half_t* rb = rbf + (size_t)e * 32;
    #pragma unroll
    for (int n = 0; n < 20; ++n) rb[n] = (half_t)(__sinf((float)(n + 1) * c) * inv);
    #pragma unroll
    for (int n = 20; n < 32; ++n) rb[n] = (half_t)0.0f;
}

// Edge message: gather phi[idx_j], Wf, v_old[idx_j]; scatter-add into s_acc/v_acc.
__global__ void edge_msg_k(const int* __restrict__ idx_i, const int* __restrict__ idx_j,
                           const float* __restrict__ phi, const half_t* __restrict__ Wf,
                           const float* __restrict__ v_old, const float* __restrict__ dirn,
                           float* __restrict__ s_acc, float* __restrict__ v_acc, int E)
{
    int tid = blockIdx.x * blockDim.x + threadIdx.x;
    int e = tid >> 7, f = tid & 127;
    if (e >= E) return;
    int i = idx_i[e], j = idx_j[e];
    size_t pb = (size_t)j * 384 + f;
    size_t wb = (size_t)e * 384 + f;
    float ds  = phi[pb]       * (float)Wf[wb];
    float dvv = phi[pb + 128] * (float)Wf[wb + 128];
    float dvs = phi[pb + 256] * (float)Wf[wb + 256];
    atomicAdd(&s_acc[(size_t)i * 128 + f], ds);
    #pragma unroll
    for (int c = 0; c < 3; ++c) {
        float dv = dvv * v_old[((size_t)j * 3 + c) * 128 + f] + dvs * dirn[3*e + c];
        atomicAdd(&v_acc[((size_t)i * 3 + c) * 128 + f], dv);
    }
}

// Update prologue: Vn = ||Vv||, dot = <Uv,Vv> over xyz; build mlp input [s, Vn] f16.
__global__ void upd_prep_k(const float* __restrict__ Uv, const float* __restrict__ Vv,
                           const float* __restrict__ s, float* __restrict__ dotUV,
                           half_t* __restrict__ xcat, int A)
{
    int tid = blockIdx.x * blockDim.x + threadIdx.x;
    int a = tid >> 7, f = tid & 127;
    if (a >= A) return;
    size_t b = (size_t)a * 3 * 128 + f;
    float u0 = Uv[b], u1 = Uv[b + 128], u2 = Uv[b + 256];
    float w0 = Vv[b], w1 = Vv[b + 128], w2 = Vv[b + 256];
    float vn = sqrtf(w0*w0 + w1*w1 + w2*w2);
    dotUV[(size_t)a * 128 + f] = u0*w0 + u1*w1 + u2*w2;
    xcat[(size_t)a * 256 + f]       = (half_t)s[(size_t)a * 128 + f];
    xcat[(size_t)a * 256 + 128 + f] = (half_t)vn;
}

// Update epilogue: v += a_vv*Uv ; s += a_ss + a_sv*dot
__global__ void upd_apply_k(const float* __restrict__ aout, const float* __restrict__ Uv,
                            const float* __restrict__ dotUV,
                            float* __restrict__ v, float* __restrict__ s, int A)
{
    int tid = blockIdx.x * blockDim.x + threadIdx.x;
    int a = tid >> 7, f = tid & 127;
    if (a >= A) return;
    size_t ab = (size_t)a * 384 + f;
    float avv = aout[ab], asv = aout[ab + 128], ass = aout[ab + 256];
    #pragma unroll
    for (int c = 0; c < 3; ++c) {
        size_t vb = ((size_t)a * 3 + c) * 128 + f;
        v[vb] += avv * Uv[vb];
    }
    s[(size_t)a * 128 + f] += ass + asv * dotUV[(size_t)a * 128 + f];
}

__global__ void zero_k(float* p, int n)
{
    int t = blockIdx.x * blockDim.x + threadIdx.x;
    if (t < n) p[t] = 0.0f;
}

// Final readout: h[a] = t_h[a,:]·lin2_w + lin2_b ; atomic pool into out[idx_m[a]].
__global__ void readout_k(const half_t* __restrict__ th, const float* __restrict__ lin2w,
                          const float* __restrict__ lin2b, const int* __restrict__ idx_m,
                          float* __restrict__ out, int A)
{
    int a = blockIdx.x * blockDim.x + threadIdx.x;
    if (a >= A) return;
    float sum = 0.0f;
    const half_t* row = th + (size_t)a * 128;
    #pragma unroll 8
    for (int k = 0; k < 128; ++k) sum += (float)row[k] * lin2w[k];
    sum += lin2b[0];
    atomicAdd(&out[idx_m[a]], sum);
}

extern "C" void kernel_launch(void* const* d_in, const int* in_sizes, int n_in,
                              void* d_out, int out_size, void* d_ws, size_t ws_size,
                              hipStream_t stream)
{
    const float* in_v   = (const float*)d_in[0];
    const int*   in_s   = (const int*)  d_in[1];
    const float* in_r   = (const float*)d_in[2];
    const int*   idx_i  = (const int*)  d_in[3];
    const int*   idx_j  = (const int*)  d_in[4];
    const int*   idx_m  = (const int*)  d_in[5];
    const float* emb    = (const float*)d_in[6];
    const float* lin1_w = (const float*)d_in[7];
    const float* lin1_b = (const float*)d_in[8];
    const float* lin2_w = (const float*)d_in[9];
    const float* lin2_b = (const float*)d_in[10];
    const float* phi1_w = (const float*)d_in[11];
    const float* phi1_b = (const float*)d_in[12];
    const float* phi2_w = (const float*)d_in[13];
    const float* phi2_b = (const float*)d_in[14];
    const float* filt_w = (const float*)d_in[15];
    const float* filt_b = (const float*)d_in[16];
    const float* U_w    = (const float*)d_in[17];
    const float* V_w    = (const float*)d_in[18];
    const float* mlp1_w = (const float*)d_in[19];
    const float* mlp1_b = (const float*)d_in[20];
    const float* mlp2_w = (const float*)d_in[21];
    const float* mlp2_b = (const float*)d_in[22];

    const int A = in_sizes[1];   // s is [A,1]
    const int E = in_sizes[3];   // idx_i is [E]
    float* out = (float*)d_out;

    // ---- workspace layout ----
    char* ws = (char*)d_ws;
    size_t off = 0;
    auto alloc = [&](size_t bytes) -> void* {
        void* p = ws + off;
        off += (bytes + 255) & ~(size_t)255;
        return p;
    };
    float*  s_cur  = (float*) alloc((size_t)A * 128 * 4);
    float*  v_cur  = (float*) alloc((size_t)A * 384 * 4);
    float*  v_old  = (float*) alloc((size_t)A * 384 * 4);
    half_t* s_h    = (half_t*)alloc((size_t)A * 128 * 2);
    half_t* v_h    = (half_t*)alloc((size_t)A * 384 * 2);
    half_t* tmp_h  = (half_t*)alloc((size_t)A * 128 * 2);
    half_t* xcat_h = (half_t*)alloc((size_t)A * 256 * 2);
    float*  dirn   = (float*) alloc((size_t)E * 3 * 4);
    float*  fcut   = (float*) alloc((size_t)E * 4);
    half_t* rbf_h  = (half_t*)alloc((size_t)E * 32 * 2);
    half_t* Wf_h   = (half_t*)alloc((size_t)E * 384 * 2);
    float*  phi    = (float*) alloc((size_t)A * 384 * 4);
    float*  Uv     = (float*) alloc((size_t)A * 384 * 4);
    float*  Vv     = (float*) alloc((size_t)A * 384 * 4);
    float*  dotUV  = (float*) alloc((size_t)A * 128 * 4);
    float*  aout   = (float*) alloc((size_t)A * 384 * 4);
    half_t* phi1T  = (half_t*)alloc(128 * 128 * 2);
    half_t* phi2T  = (half_t*)alloc(384 * 128 * 2);
    half_t* filtT  = (half_t*)alloc(384 * 32 * 2);
    half_t* UT     = (half_t*)alloc(128 * 128 * 2);
    half_t* VT     = (half_t*)alloc(128 * 128 * 2);
    half_t* mlp1T  = (half_t*)alloc(128 * 256 * 2);
    half_t* mlp2T  = (half_t*)alloc(384 * 128 * 2);
    half_t* lin1T  = (half_t*)alloc(128 * 128 * 2);
    (void)ws_size; (void)n_in;

    auto gemm = [&](const half_t* Ah, const half_t* BT, const float* bias,
                    const float* rsc, float* oF, half_t* oH,
                    int Mr, int Nc, int Kd, int act) {
        dim3 grid(Nc / 64, (Mr / 16 + 7) / 8);
        if (Kd == 32)
            gemm16_wmma<32><<<grid, 256, 0, stream>>>(Ah, BT, bias, rsc, oF, oH,
                                                      Mr, Nc, act);
        else if (Kd == 128)
            gemm16_wmma<128><<<grid, 256, 0, stream>>>(Ah, BT, bias, rsc, oF, oH,
                                                       Mr, Nc, act);
        else
            gemm16_wmma<256><<<grid, 256, 0, stream>>>(Ah, BT, bias, rsc, oF, oH,
                                                       Mr, Nc, act);
    };
    auto pack = [&](const float* W, half_t* WT, int K, int N, int Kpad) {
        int total = N * Kpad;
        pack_bt_k<<<(total + 255) / 256, 256, 0, stream>>>(W, WT, K, N, Kpad);
    };
    auto f2h = [&](const float* x, half_t* y, size_t n) {
        f2h_k<<<(unsigned)((n + 255) / 256), 256, 0, stream>>>(x, y, n);
    };

    // ---- one-time prep: weight packing, embedding, edge geometry, Wf ----
    pack(phi1_w, phi1T, 128, 128, 128);
    pack(phi2_w, phi2T, 128, 384, 128);
    pack(filt_w, filtT,  20, 384,  32);
    pack(U_w,    UT,    128, 128, 128);
    pack(V_w,    VT,    128, 128, 128);
    pack(mlp1_w, mlp1T, 256, 128, 256);
    pack(mlp2_w, mlp2T, 128, 384, 128);
    pack(lin1_w, lin1T, 128, 128, 128);

    embed_k<<<(A * 128 + 255) / 256, 256, 0, stream>>>(in_s, emb, s_cur, A);
    hipMemcpyAsync(v_cur, in_v, (size_t)A * 384 * 4, hipMemcpyDeviceToDevice, stream);

    edge_prep_k<<<(E + 255) / 256, 256, 0, stream>>>(in_r, dirn, fcut, rbf_h, E);
    // Wf = (rbf @ filt_w + filt_b) * fcut  — r is loop-invariant, compute once.
    gemm(rbf_h, filtT, filt_b, fcut, nullptr, Wf_h, E, 384, 32, 0);

    // ---- 3 message/update iterations (tied weights) ----
    for (int it = 0; it < 3; ++it) {
        // message
        f2h(s_cur, s_h, (size_t)A * 128);
        gemm(s_h,   phi1T, phi1_b, nullptr, nullptr, tmp_h, A, 128, 128, 1); // silu
        gemm(tmp_h, phi2T, phi2_b, nullptr, phi,     nullptr, A, 384, 128, 0);
        hipMemcpyAsync(v_old, v_cur, (size_t)A * 384 * 4, hipMemcpyDeviceToDevice, stream);
        edge_msg_k<<<(E * 128) / 256, 256, 0, stream>>>(idx_i, idx_j, phi, Wf_h,
                                                        v_old, dirn, s_cur, v_cur, E);
        // update
        f2h(v_cur, v_h, (size_t)A * 384);
        gemm(v_h, UT, nullptr, nullptr, Uv, nullptr, 3 * A, 128, 128, 0);
        gemm(v_h, VT, nullptr, nullptr, Vv, nullptr, 3 * A, 128, 128, 0);
        upd_prep_k<<<(A * 128 + 255) / 256, 256, 0, stream>>>(Uv, Vv, s_cur, dotUV,
                                                              xcat_h, A);
        gemm(xcat_h, mlp1T, mlp1_b, nullptr, nullptr, tmp_h, A, 128, 256, 1); // silu
        gemm(tmp_h,  mlp2T, mlp2_b, nullptr, aout,    nullptr, A, 384, 128, 0);
        upd_apply_k<<<(A * 128 + 255) / 256, 256, 0, stream>>>(aout, Uv, dotUV,
                                                               v_cur, s_cur, A);
    }

    // ---- readout ----
    f2h(s_cur, s_h, (size_t)A * 128);
    gemm(s_h, lin1T, lin1_b, nullptr, nullptr, tmp_h, A, 128, 128, 1); // silu
    zero_k<<<(out_size + 255) / 256, 256, 0, stream>>>(out, out_size);
    readout_k<<<(A + 255) / 256, 256, 0, stream>>>(tmp_h, lin2_w, lin2_b, idx_m, out, A);
}
